// OurModel_18451179503960
// MI455X (gfx1250) — matrix operand order repaired and checked by reading, not compiled, not run on hardware
//
#include <hip/hip_runtime.h>
#include <hip/hip_bf16.h>
#include <math.h>

// ---------------- types for WMMA ----------------
typedef __bf16 bf16_t;
typedef bf16_t v16bf __attribute__((ext_vector_type(16)));
typedef bf16_t v8bf  __attribute__((ext_vector_type(8)));
typedef float  v8f   __attribute__((ext_vector_type(8)));

// ---------------- problem constants ----------------
#define NN     20000
#define EE     320000
#define F_IN   512
#define H1_    1024
#define H2_    512
#define F_OUT  128
#define D1_    152
#define D2_    48

// ======================================================================
// elementwise: f32 -> bf16
__global__ void k_cvt_bf16(const float* __restrict__ in, bf16_t* __restrict__ out, size_t n) {
    size_t i = (size_t)blockIdx.x * blockDim.x + threadIdx.x;
    if (i < n) out[i] = (bf16_t)in[i];
}

// transpose + convert: W (K x M row-major, f32) -> Wt (M x K row-major, bf16)
__global__ void k_wt_cvt(const float* __restrict__ W, bf16_t* __restrict__ Wt, int K, int M) {
    size_t i = (size_t)blockIdx.x * blockDim.x + threadIdx.x;
    if (i >= (size_t)K * M) return;
    int k = (int)(i / M);
    int m = (int)(i % M);
    Wt[(size_t)m * K + k] = (bf16_t)W[i];
}

// fill with broadcast bias (or zero). F is a power of two here.
__global__ void k_fill(float* __restrict__ out, const float* __restrict__ b, int F, size_t total) {
    size_t i = (size_t)blockIdx.x * blockDim.x + threadIdx.x;
    if (i < total) out[i] = b ? b[i & (size_t)(F - 1)] : 0.0f;
}

// activation: out = bf16(tanh(in [+ bias]))
__global__ void k_act_tanh(const float* __restrict__ in, const float* __restrict__ b,
                           bf16_t* __restrict__ out, int F, size_t total) {
    size_t i = (size_t)blockIdx.x * blockDim.x + threadIdx.x;
    if (i >= total) return;
    float v = in[i];
    if (b) v += b[i & (size_t)(F - 1)];
    out[i] = (bf16_t)tanhf(v);
}

// ======================================================================
// SpMM scatter: one block per edge; out[dst[e]] += w[e] * in[src[e]]
__global__ void k_spmm(const int* __restrict__ src, const int* __restrict__ dst,
                       const float* __restrict__ ew, const float* __restrict__ hin,
                       float* __restrict__ out, int F4) {
    int e = blockIdx.x;
    int s = src[e];
    int d = dst[e];
    float w = ew[e];
    const float4* in4 = (const float4*)hin + (size_t)s * F4;
    float* op = out + (size_t)d * ((size_t)F4 * 4);
    for (int f = threadIdx.x; f < F4; f += blockDim.x) {
        float4 v = in4[f];
        unsafeAtomicAdd(op + (size_t)f * 4 + 0, w * v.x);
        unsafeAtomicAdd(op + (size_t)f * 4 + 1, w * v.y);
        unsafeAtomicAdd(op + (size_t)f * 4 + 2, w * v.z);
        unsafeAtomicAdd(op + (size_t)f * 4 + 3, w * v.w);
    }
}

// ======================================================================
// Register-blocked bf16 WMMA GEMM: C[M x Ncols] = A[M x K] * Bt[Ncols x K]^T
// block = 256 threads = 8 waves; each wave computes an MT x NT grid of
// 16x16 tiles (fragment reuse: MT+NT fragment loads -> MT*NT WMMAs).
// grid.x = M/(16*MT), grid.y = Ncols/(8*16*NT). All dims exact multiples
// (no masking; EXEC stays all-ones as WMMA requires).
template <int MT, int NT>
__global__ void __launch_bounds__(256)
k_gemm_bf16_wmma(const bf16_t* __restrict__ A, const bf16_t* __restrict__ Bt,
                 float* __restrict__ C, int K, int Ncols) {
    const int lane  = threadIdx.x & 31;
    const int wave  = threadIdx.x >> 5;
    const int mBase = blockIdx.x * (16 * MT);
    const int nBase = (blockIdx.y * 8 + wave) * (16 * NT);
    const int half  = lane >> 4;   // 0: lanes 0-15, 1: lanes 16-31
    const int l15   = lane & 15;

    const bf16_t* aRow[MT];
    const bf16_t* bRow[NT];
#pragma unroll
    for (int mt = 0; mt < MT; ++mt)
        aRow[mt] = A + (size_t)(mBase + mt * 16 + l15) * K;   // A row (M = l15)
#pragma unroll
    for (int nt = 0; nt < NT; ++nt)
        bRow[nt] = Bt + (size_t)(nBase + nt * 16 + l15) * K;  // B column (N = l15)

    v8f acc[MT][NT] = {};

    for (int k = 0; k < K; k += 32) {
        v16bf av[MT], bv[NT];
#pragma unroll
        for (int mt = 0; mt < MT; ++mt) {
            // A fragment (16-bit A 16x32 layout): half 0 -> K {0..7,16..23},
            //                                     half 1 -> K {8..15,24..31}
            v8bf a0 = *(const v8bf*)(aRow[mt] + k + half * 8);
            v8bf a1 = *(const v8bf*)(aRow[mt] + k + 16 + half * 8);
#pragma unroll
            for (int i = 0; i < 8; ++i) { av[mt][i] = a0[i]; av[mt][i + 8] = a1[i]; }
        }
#pragma unroll
        for (int nt = 0; nt < NT; ++nt) {
            // B fragment: column-per-lane, same K order as A:
            // half 0 -> K {0..7,16..23}, half 1 -> K {8..15,24..31}
            v8bf b0 = *(const v8bf*)(bRow[nt] + k + half * 8);
            v8bf b1 = *(const v8bf*)(bRow[nt] + k + 16 + half * 8);
#pragma unroll
            for (int i = 0; i < 8; ++i) { bv[nt][i] = b0[i]; bv[nt][i + 8] = b1[i]; }
        }
#pragma unroll
        for (int mt = 0; mt < MT; ++mt)
#pragma unroll
            for (int nt = 0; nt < NT; ++nt)
                acc[mt][nt] = __builtin_amdgcn_wmma_f32_16x16x32_bf16(
                    /*neg_a=*/false, av[mt], /*neg_b=*/false, bv[nt],
                    /*c_mod=*/(short)0, acc[mt][nt],
                    /*reuse_a=*/false, /*reuse_b=*/false);
    }

    // C/D layout: VGPR r -> lanes 0-15: (M=r, N=lane); lanes 16-31: (M=8+r, N=lane-16)
#pragma unroll
    for (int mt = 0; mt < MT; ++mt)
#pragma unroll
        for (int nt = 0; nt < NT; ++nt) {
            float* cPtr = C + (size_t)(mBase + mt * 16 + half * 8) * Ncols
                            + nBase + nt * 16 + l15;
#pragma unroll
            for (int r = 0; r < 8; ++r) cPtr[(size_t)r * Ncols] = acc[mt][nt][r];
        }
}

// ======================================================================
// small dense layer: out[n][j] = act(sum_k in[n][k] * W[k][j] + b[j])
__global__ void k_dense(const float* __restrict__ in, const float* __restrict__ W,
                        const float* __restrict__ b, float* __restrict__ out,
                        int K, int J, int n, int doRelu) {
    size_t i = (size_t)blockIdx.x * blockDim.x + threadIdx.x;
    if (i >= (size_t)n * J) return;
    int node = (int)(i / J);
    int j    = (int)(i % J);
    const float* ip = in + (size_t)node * K;
    float acc = b[j];
    for (int k = 0; k < K; ++k) acc += ip[k] * W[(size_t)k * J + j];
    if (doRelu && acc < 0.0f) acc = 0.0f;
    out[i] = acc;
}

// ======================================================================
static inline unsigned cdiv(size_t a, unsigned b) { return (unsigned)((a + b - 1) / b); }

extern "C" void kernel_launch(void* const* d_in, const int* in_sizes, int n_in,
                              void* d_out, int out_size, void* d_ws, size_t ws_size,
                              hipStream_t stream) {
    const float* x   = (const float*)d_in[0];
    const int*   src = (const int*)  d_in[1];
    const int*   dst = (const int*)  d_in[2];
    const float* ew  = (const float*)d_in[3];
    const float* W1  = (const float*)d_in[4];
    const float* b1  = (const float*)d_in[5];
    const float* W2  = (const float*)d_in[6];
    const float* b2  = (const float*)d_in[7];
    const float* W3  = (const float*)d_in[8];
    const float* b3  = (const float*)d_in[9];
    const float* Wd1 = (const float*)d_in[10];
    const float* bd1 = (const float*)d_in[11];
    const float* Wd2 = (const float*)d_in[12];
    const float* bd2 = (const float*)d_in[13];
    const float* Wd3 = (const float*)d_in[14];
    const float* bd3 = (const float*)d_in[15];
    float* out = (float*)d_out;

    // workspace layout (bytes)
    char* ws = (char*)d_ws;
    bf16_t* ABF = (bf16_t*)(ws);                              // up to 20000*1024 bf16 = 41 MB
    bf16_t* WBF = (bf16_t*)(ws + (size_t)41 * 1024 * 1024);   // up to 1024*1024 bf16 = 2 MB
    float*  TMP = (float*)(ws + (size_t)44 * 1024 * 1024);    // up to 20000*1024 f32 = 82 MB
    float*  ACC = (float*)(ws + (size_t)128 * 1024 * 1024);   // up to 20000*1024 f32 = 82 MB
    float*  D1B = TMP;                                        // reuse after layer 3
    float*  D2B = (float*)(ws + (size_t)60 * 1024 * 1024);    // inside TMP region, past D1B

    const unsigned BLK = 256;

    // ---------------- Layer 1 (reassociated: g1 = (A @ x) @ W1) ----------------
    {   // ACC = 0 (N x F_IN)
        size_t tot = (size_t)NN * F_IN;
        k_fill<<<cdiv(tot, BLK), BLK, 0, stream>>>(ACC, nullptr, F_IN, tot);
        // ACC += scatter(x)
        k_spmm<<<EE, F_IN / 4, 0, stream>>>(src, dst, ew, x, ACC, F_IN / 4);
        // ABF = bf16(ACC)
        k_cvt_bf16<<<cdiv(tot, BLK), BLK, 0, stream>>>(ACC, ABF, tot);
        // WBF = W1^T (H1 x F_IN)
        k_wt_cvt<<<cdiv((size_t)F_IN * H1_, BLK), BLK, 0, stream>>>(W1, WBF, F_IN, H1_);
        // TMP = ABF @ W1  (N x H1)
        dim3 g(NN / 32, H1_ / 256);
        k_gemm_bf16_wmma<2, 2><<<g, BLK, 0, stream>>>(ABF, WBF, TMP, F_IN, H1_);
        // ABF = bf16(tanh(TMP + b1))  (N x H1)
        size_t tot1 = (size_t)NN * H1_;
        k_act_tanh<<<cdiv(tot1, BLK), BLK, 0, stream>>>(TMP, b1, ABF, H1_, tot1);
    }

    // ---------------- Layer 2: h2 = tanh(A @ (h1 @ W2) + b2) ----------------
    {
        k_wt_cvt<<<cdiv((size_t)H1_ * H2_, BLK), BLK, 0, stream>>>(W2, WBF, H1_, H2_);
        dim3 g(NN / 32, H2_ / 256);
        k_gemm_bf16_wmma<2, 2><<<g, BLK, 0, stream>>>(ABF, WBF, TMP, H1_, H2_);
        size_t tot = (size_t)NN * H2_;
        k_fill<<<cdiv(tot, BLK), BLK, 0, stream>>>(ACC, b2, H2_, tot);
        k_spmm<<<EE, H2_ / 4, 0, stream>>>(src, dst, ew, TMP, ACC, H2_ / 4);
        k_act_tanh<<<cdiv(tot, BLK), BLK, 0, stream>>>(ACC, nullptr, ABF, H2_, tot);
    }

    // ---------------- Layer 3: g = A @ (h2 @ W3) + b3 ----------------
    {
        k_wt_cvt<<<cdiv((size_t)H2_ * F_OUT, BLK), BLK, 0, stream>>>(W3, WBF, H2_, F_OUT);
        dim3 g(NN / 32, F_OUT / 128);
        k_gemm_bf16_wmma<2, 1><<<g, BLK, 0, stream>>>(ABF, WBF, TMP, H2_, F_OUT);
        size_t tot = (size_t)NN * F_OUT;
        k_fill<<<cdiv(tot, BLK), BLK, 0, stream>>>(ACC, b3, F_OUT, tot);
        k_spmm<<<EE, F_OUT / 4, 0, stream>>>(src, dst, ew, TMP, ACC, F_OUT / 4);
        // g lives in ACC (N x F_OUT, f32)
    }

    // ---------------- Dense head ----------------
    {
        size_t t1 = (size_t)NN * D1_;
        k_dense<<<cdiv(t1, BLK), BLK, 0, stream>>>(ACC, Wd1, bd1, D1B, F_OUT, D1_, NN, 1);
        size_t t2 = (size_t)NN * D2_;
        k_dense<<<cdiv(t2, BLK), BLK, 0, stream>>>(D1B, Wd2, bd2, D2B, D1_, D2_, NN, 1);
        size_t t3 = (size_t)NN;
        k_dense<<<cdiv(t3, BLK), BLK, 0, stream>>>(D2B, Wd3, bd3, out, D2_, 1, NN, 0);
    }
}
